// Memory_64158221467746
// MI455X (gfx1250) — matrix operand order repaired and checked by reading, not compiled, not run on hardware
//
#include <hip/hip_runtime.h>
#include <hip/hip_bf16.h>

typedef __attribute__((ext_vector_type(16))) _Float16 v16h;
typedef __attribute__((ext_vector_type(8)))  _Float16 v8h;
typedef __attribute__((ext_vector_type(8)))  float    v8f;

// Problem sizes (from reference)
constexpr int SEQ  = 512;
constexpr int B    = 64;
constexpr int NIN  = 1024;
constexpr int NHID = 1024;
constexpr int NG   = 4 * NHID;      // 4096 gate columns
constexpr int KH   = 1024;          // K per half (x-half / h-half)
constexpr int WROW = NIN + NHID;    // 2048, W row stride
constexpr int ROWP = 40;            // LDS A-row pitch in halves (80 B):
                                    // 16B-aligned rows, bank-conflict-free reads

// ---------------------------------------------------------------------------
// f32 -> f16 conversion (4 elements/thread)
// ---------------------------------------------------------------------------
__global__ void cvt_f32_f16(const float* __restrict__ src,
                            _Float16* __restrict__ dst, int n) {
  int i = (blockIdx.x * blockDim.x + threadIdx.x) * 4;
  if (i + 3 < n) {
    float4 v = *(const float4*)(src + i);
    dst[i + 0] = (_Float16)v.x;
    dst[i + 1] = (_Float16)v.y;
    dst[i + 2] = (_Float16)v.z;
    dst[i + 3] = (_Float16)v.w;
  }
}

// zero c (f32) and h0 (f16)
__global__ void init_state(float* __restrict__ c, _Float16* __restrict__ h0, int n) {
  int i = blockIdx.x * blockDim.x + threadIdx.x;
  if (i < n) {
    c[i]  = 0.0f;
    h0[i] = (_Float16)0.0f;
  }
}

__global__ void copy_f32(const float* __restrict__ src, float* __restrict__ dst, int n) {
  int i = blockIdx.x * blockDim.x + threadIdx.x;
  if (i < n) dst[i] = src[i];
}

__device__ __forceinline__ float sigmoidf(float x) {
  return 1.0f / (1.0f + __expf(-x));
}

// Low 32 bits of a generic pointer to a __shared__ object == wave-relative LDS
// byte address (ISA 10.2); that is what the async-load
// VDST operand expects (added to the LDS base and the instruction offset).
__device__ __forceinline__ uint32_t lds_addr32(const void* p) {
  return (uint32_t)(uintptr_t)p;
}

// CDNA5 async copy: 16 bytes global -> LDS per lane, tracked by ASYNCcnt.
#define ASYNC_G2L_B128(ldsoff_u32, gptr)                                  \
  asm volatile("global_load_async_to_lds_b128 %0, %1, off"                \
               :: "v"(ldsoff_u32), "v"(gptr) : "memory")

#define WAIT_ASYNC0() asm volatile("s_wait_asynccnt 0" ::: "memory")

// Raw workgroup barrier WITHOUT the conservative loadcnt/dscnt flush that
// __syncthreads() inserts (which would serialize the B-fragment prefetch).
// Safe here: async fills are waited explicitly, and all LDS reads of the
// buffer being refilled were already consumed by WMMAs (in-order DS returns
// mean the compiler's s_wait_dscnt before the last WMMA implies dscnt==0).
#define WG_BARRIER() \
  asm volatile("s_barrier_signal -1\n\ts_barrier_wait -1" ::: "memory")

// Keep each K-chunk's loads+WMMAs self-contained so the scheduler doesn't
// rotate the 32 accumulator VGPRs (v_mov_b64 + WMMA-hazard v_nop churn).
#if __has_builtin(__builtin_amdgcn_sched_barrier)
#define SCHED_FENCE() __builtin_amdgcn_sched_barrier(0)
#else
#define SCHED_FENCE() asm volatile("" ::: "memory")
#endif

#if __has_builtin(__builtin_amdgcn_sched_group_barrier)
#define HAVE_SGB 1
#else
#define HAVE_SGB 0
#endif

// ---------------------------------------------------------------------------
// One LSTM timestep.
//   grid = 64 workgroups, each owns 16 hidden units x 64 batch x 4 gates.
//   wave w (of 8): gate = w%4, K-half = w/4 (0: x part, 1: h part).
//   Each wave: 4 WMMA C-tiles (M=64 as 4x16, N=16), K=1024 in steps of 32.
//   A tiles: double-buffered async global->LDS (static buffer indices).
//   B fragments: software-pipelined one K-step ahead in registers.
// ---------------------------------------------------------------------------
__global__ __launch_bounds__(256)
void lstm_step(const _Float16* __restrict__ x_t,    // [64][1024] f16
               const _Float16* __restrict__ h_in,   // [64][1024] f16
               const _Float16* __restrict__ W16,    // [4096][2048] f16
               const float*    __restrict__ bias,   // [4096]
               float* __restrict__ c,               // [64][1024] f32 (state)
               _Float16* __restrict__ h_out16,      // [64][1024] f16
               float* __restrict__ out_t)           // [64][1024] f32
{
  // [buf][xh][64 rows * ROWP halves] : 2 x 2 x 5 KB = 20 KB
  __shared__ _Float16 ldsA[2][2][64 * ROWP];
  __shared__ float    pre[8][64 * 16];   // per-wave partial preactivations (32 KB)

  const int tid   = threadIdx.x;
  const int wave  = tid >> 5;
  const int lane  = tid & 31;
  const int g     = lane >> 4;        // half-wave group (WMMA layout)
  const int n     = lane & 15;        // N column / M row selector
  const int gate  = wave & 3;
  const int khalf = wave >> 2;        // 0 -> x columns of W, 1 -> h columns
  const int nbase = blockIdx.x * 16;  // hidden-unit tile base

  v8f acc[4] = {};                    // 4 M-tiles of 16x16 f32 accumulators

  // W row this wave's lane column maps to: j = gate*NHID + nbase + n
  const _Float16* wrow =
      W16 + (size_t)(gate * NHID + nbase + n) * WROW + (size_t)khalf * KH;

  // cooperative A-chunk staging: each thread async-copies 8 halves per chunk
  const int cm = tid >> 2;            // row 0..63
  const int cc = (tid & 3) * 8;       // col 0..24 step 8
  const _Float16* xsrc = x_t  + (size_t)cm * NIN  + cc;
  const _Float16* hsrc = h_in + (size_t)cm * NHID + cc;
  const uint32_t ldx[2] = { lds_addr32(&ldsA[0][0][cm * ROWP + cc]),
                            lds_addr32(&ldsA[1][0][cm * ROWP + cc]) };
  const uint32_t ldh[2] = { lds_addr32(&ldsA[0][1][cm * ROWP + cc]),
                            lds_addr32(&ldsA[1][1][cm * ROWP + cc]) };

  auto stage = [&](int bufsel, int kc) {
    ASYNC_G2L_B128(ldx[bufsel], xsrc + kc);
    ASYNC_G2L_B128(ldh[bufsel], hsrc + kc);
  };

  // current B fragment (32x16 f16): b[e] = W[j][khalf*KH + kc + g*16 + e]
  v16h bcur = *(const v16h*)(wrow + g * 16);

  const _Float16* A0 = ldsA[0][khalf];
  const _Float16* A1 = ldsA[1][khalf];

  auto compute = [&](const _Float16* Achunk, int kcb, bool pfB) {
    v16h bn = {};
    if (pfB) bn = *(const v16h*)(wrow + kcb + 32 + g * 16);
    // A fragments (16x32 f16): row m = mt*16 + lane%16,
    //   a[e] = A[m][g*16 + e]
    v16h af[4];
#pragma unroll
    for (int mt = 0; mt < 4; ++mt) {
      const _Float16* arow = Achunk + (mt * 16 + n) * ROWP;
      v8h lo = *(const v8h*)(arow + g * 16);
      v8h hi = *(const v8h*)(arow + g * 16 + 8);
      af[mt] = __builtin_shufflevector(lo, hi,
          0, 1, 2, 3, 4, 5, 6, 7, 8, 9, 10, 11, 12, 13, 14, 15);
    }
#pragma unroll
    for (int mt = 0; mt < 4; ++mt)
      acc[mt] = __builtin_amdgcn_wmma_f32_16x16x32_f16(
          false, af[mt], false, bcur, (short)0, acc[mt], false, false);
    if (pfB) bcur = bn;
#if HAVE_SGB
    // Pin per-chunk issue order: B prefetch first, then overlap the two
    // 4-DS-read fragment groups with the WMMA stream (bounded pressure, no
    // full s_wait_dscnt 0 round-trip per WMMA).
    if (pfB) __builtin_amdgcn_sched_group_barrier(0x020, 2, 0); // VMEM reads
    __builtin_amdgcn_sched_group_barrier(0x100, 4, 0);          // DS reads
    __builtin_amdgcn_sched_group_barrier(0x008, 1, 0);          // WMMA
    __builtin_amdgcn_sched_group_barrier(0x100, 4, 0);          // DS reads
    __builtin_amdgcn_sched_group_barrier(0x008, 3, 0);          // WMMA x3
#endif
  };

  // prefetch first chunk into buffer 0
  stage(0, 0);

  // double-pumped main loop: all buffer indices / guards compile-time constant
  for (int kc = 0; kc < KH - 64; kc += 64) {
    WAIT_ASYNC0();        // this wave's async fills landed in LDS
    WG_BARRIER();         // all waves' fills landed; prior-buffer reads done
    stage(1, kc + 32);
    compute(A0, kc, true);
    SCHED_FENCE();

    WAIT_ASYNC0();
    WG_BARRIER();
    stage(0, kc + 64);
    compute(A1, kc + 32, true);
    SCHED_FENCE();
  }
  // peeled tail: kc = KH-64, KH-32
  WAIT_ASYNC0();
  WG_BARRIER();
  stage(1, KH - 32);
  compute(A0, KH - 64, true);
  SCHED_FENCE();

  WAIT_ASYNC0();
  WG_BARRIER();
  compute(A1, KH - 32, false);

  // C/D layout: acc[mt][r] = D[m = mt*16 + r + 8*g][n = lane%16]
#pragma unroll
  for (int mt = 0; mt < 4; ++mt)
#pragma unroll
    for (int r = 0; r < 8; ++r)
      pre[wave][(mt * 16 + r + 8 * g) * 16 + n] = acc[mt][r];
  __syncthreads();   // real flush: pre[] stores must be visible to all waves

  // Fused LSTM cell: 1024 cells (64 batch x 16 hid), 4 per thread
#pragma unroll
  for (int e = 0; e < 4; ++e) {
    int cell = tid + e * 256;
    int m  = cell >> 4;
    int nl = cell & 15;
    int ng = nbase + nl;
    int pi = m * 16 + nl;
    float ai  = pre[0][pi] + pre[4][pi] + bias[0 * NHID + ng];
    float af_ = pre[1][pi] + pre[5][pi] + bias[1 * NHID + ng];
    float ao  = pre[2][pi] + pre[6][pi] + bias[2 * NHID + ng];
    float ag  = pre[3][pi] + pre[7][pi] + bias[3 * NHID + ng];
    float iv = sigmoidf(ai);
    float fv = sigmoidf(af_);
    float ov = sigmoidf(ao);
    float gv = tanhf(ag);
    size_t ci = (size_t)m * NHID + ng;
    float cv = c[ci] * fv + iv * gv;
    c[ci] = cv;
    float hv = ov * tanhf(cv);
    out_t[ci]   = hv;
    h_out16[ci] = (_Float16)hv;
  }
}

// ---------------------------------------------------------------------------
extern "C" void kernel_launch(void* const* d_in, const int* in_sizes, int n_in,
                              void* d_out, int out_size, void* d_ws, size_t ws_size,
                              hipStream_t stream) {
  (void)in_sizes; (void)n_in; (void)out_size; (void)ws_size;

  const float* x  = (const float*)d_in[0];   // [512][64][1024]
  const float* W  = (const float*)d_in[1];   // [4096][2048]
  const float* bi = (const float*)d_in[2];   // [4096]
  float* out = (float*)d_out;                // [512*64*1024 output] + [64*1024 h]

  // workspace layout
  char* ws = (char*)d_ws;
  _Float16* W16 = (_Float16*)ws;                                   // 16 MB
  _Float16* x16 = (_Float16*)(ws + (size_t)16777216);              // 64 MB
  float*    c   = (float*)   (ws + (size_t)16777216 + 67108864);   // 256 KB
  _Float16* hA  = (_Float16*)(ws + (size_t)16777216 + 67108864 + 262144);
  _Float16* hB  = hA + (size_t)B * NHID;

  // per-launch conversions, fully deterministic
  {
    int nW = NG * WROW;                 // 8,388,608
    cvt_f32_f16<<<nW / 4 / 256, 256, 0, stream>>>(W, W16, nW);
    int nX = SEQ * B * NIN;             // 33,554,432
    cvt_f32_f16<<<nX / 4 / 256, 256, 0, stream>>>(x, x16, nX);
    int nS = B * NHID;                  // 65,536
    init_state<<<nS / 256, 256, 0, stream>>>(c, hA, nS);
  }

  // sequential recurrence: 512 step kernels
  for (int t = 0; t < SEQ; ++t) {
    const _Float16* hin = (t & 1) ? hB : hA;
    _Float16*       hout = (t & 1) ? hA : hB;
    lstm_step<<<NHID / 16, 256, 0, stream>>>(
        x16 + (size_t)t * B * NIN, hin, W16, bi, c, hout,
        out + (size_t)t * B * NHID);
  }

  // final h = output[SEQ-1] -> second tuple element
  {
    int nS = B * NHID;
    copy_f32<<<nS / 256, 256, 0, stream>>>(
        out + (size_t)(SEQ - 1) * B * NHID,
        out + (size_t)SEQ * B * NHID, nS);
  }
}
